// LongRangeDW_49813030699047
// MI455X (gfx1250) — matrix operand order repaired and checked by reading, not compiled, not run on hardware
//
#include <hip/hip_runtime.h>

// ---------------- problem constants ----------------
#define BB   8
#define CC   128
#define HH   128
#define WW   128
#define C5   640          // 5*C
#define TH   4            // tile height
#define TW   16           // tile width
#define NPT  64           // points per tile = TH*TW
#define HALO 13           // dil12 (12) + stacking shift (1)
#define PH   (TH + 2*HALO)    // 30
#define PW   (TW + 2*HALO)    // 42
#define PATCH (PH*PW)         // 1260
#define YSTR 648              // 640 + 8 pad (bf16 elems) -> 1296B rows, 16B aligned

typedef __attribute__((ext_vector_type(8)))  float        v8f;
typedef __attribute__((ext_vector_type(16))) __bf16       v16bf;
typedef __attribute__((ext_vector_type(4)))  unsigned int v4u;
typedef __attribute__((ext_vector_type(8)))  unsigned int v8u;

// ---------------- w4 f32 -> bf16 convert ----------------
__global__ void lrdw_cvt_w4_bf16(const float* __restrict__ w, __bf16* __restrict__ o, int n) {
    int i = blockIdx.x * 256 + threadIdx.x;
    if (i < n) o[i] = (__bf16)w[i];
}

// 19-tap depthwise value for one (channel, stacking-shift, point).
// patch is zero-padded x (handles the outer/stacking pad); the inner
// (pre-shift) bound is checked explicitly against global coords.
__device__ __forceinline__ float lrdw_tap19(const float* __restrict__ patch,
                                            const float* __restrict__ wrow,
                                            int li, int lj, int dh, int dw,
                                            int gi, int gj) {
    float acc = wrow[19];                                   // b1+b2+b3 fused
    acc += wrow[0] * patch[(li + dh) * PW + (lj + dw)];     // 1x1 branch
#pragma unroll
    for (int u = -1; u <= 1; ++u) {
#pragma unroll
        for (int v = -1; v <= 1; ++v) {
            const int wi = (u + 1) * 3 + (v + 1);
            const bool ok8  = ((unsigned)(gi + 8*u)  < (unsigned)HH) &&
                              ((unsigned)(gj + 8*v)  < (unsigned)WW);
            const bool ok12 = ((unsigned)(gi + 12*u) < (unsigned)HH) &&
                              ((unsigned)(gj + 12*v) < (unsigned)WW);
            float s8  = ok8  ? patch[(li + 8*u  + dh) * PW + (lj + 8*v  + dw)] : 0.0f;
            float s12 = ok12 ? patch[(li + 12*u + dh) * PW + (lj + 12*v + dw)] : 0.0f;
            acc += wrow[1  + wi] * s8;
            acc += wrow[10 + wi] * s12;
        }
    }
    return acc;
}

// ---------------- fused stencil + WMMA pointwise GEMM ----------------
__global__ __launch_bounds__(256)
void lrdw_fused_kernel(const float* __restrict__ x,
                       const float* __restrict__ w1, const float* __restrict__ b1,
                       const float* __restrict__ w2, const float* __restrict__ b2,
                       const float* __restrict__ w3, const float* __restrict__ b3,
                       const float* __restrict__ b4,
                       const __bf16* __restrict__ w4b,   // [128][640] bf16
                       float* __restrict__ out) {
    __shared__ __align__(16) float  wtab[C5 * 20];        // 50 KB
    __shared__ __align__(16) float  patch[2][PATCH];      // 10 KB (double buffered)
    __shared__ __align__(16) __bf16 yt[NPT * YSTR];       // 81 KB, y transposed [p][c5]

    const int t  = threadIdx.x;
    const int bb = blockIdx.z;
    const int i0 = blockIdx.y * TH;
    const int j0 = blockIdx.x * TW;

    // ---- stage depthwise weight table: [c5][ w1 | w2*9 | w3*9 | b1+b2+b3 ] ----
    for (int idx = t; idx < C5 * 20; idx += 256) {
        const int c5 = idx / 20, s = idx % 20;
        float v;
        if      (s == 0)  v = w1[c5];
        else if (s < 10)  v = w2[c5 * 9 + (s - 1)];
        else if (s < 19)  v = w3[c5 * 9 + (s - 10)];
        else              v = b1[c5] + b2[c5] + b3[c5];
        wtab[idx] = v;
    }

    // per-thread fixed (k, point) assignment; threads 0..63 also do k=4
    const int p  = t & 63;
    const int km = t >> 6;                     // 0..3
    const int pi = p >> 4, pj = p & 15;
    const int gi = i0 + pi, gj = j0 + pj;
    const int li = HALO + pi, lj = HALO + pj;
    const int shH[5] = { 1, -1, 0, 0, 0 };     // below, above, right, left, center
    const int shW[5] = { 0,  0, 1, -1, 0 };
    const int dh = shH[km], dw = shW[km];

    // ---- producer: per-channel patch load + 19-tap stencil -> bf16 y tile ----
    for (int c = 0; c < CC; ++c) {
        float* pb = patch[c & 1];
        const float* __restrict__ xc = x + ((size_t)(bb * CC + c)) * (HH * WW);
        for (int idx = t; idx < PATCH; idx += 256) {
            const int ppi = idx / PW, ppj = idx % PW;
            const int gpi = i0 - HALO + ppi, gpj = j0 - HALO + ppj;
            const bool ok = ((unsigned)gpi < (unsigned)HH) && ((unsigned)gpj < (unsigned)WW);
            pb[idx] = ok ? xc[gpi * WW + gpj] : 0.0f;
        }
        __syncthreads();   // fill->compute; also fences compute(c-1) vs fill(c+1)

        {
            const int c5 = km * CC + c;
            const float v = lrdw_tap19(pb, &wtab[c5 * 20], li, lj, dh, dw, gi, gj);
            yt[p * YSTR + c5] = (__bf16)v;
        }
        if (t < 64) {                           // 5th (center) stacking slice
            const int c5 = 4 * CC + c;
            const float v = lrdw_tap19(pb, &wtab[c5 * 20], li, lj, 0, 0, gi, gj);
            yt[p * YSTR + c5] = (__bf16)v;
        }
    }
    __syncthreads();

    // ---- GEMM: out(128 x 64) = w4(128 x 640) * y(640 x 64), bf16 WMMA ----
    const int wave  = t >> 5;
    const int lane  = t & 31;
    const int l15   = lane & 15;
    const int lhalf = lane >> 4;
    const int m0    = wave * 16;
    const int arow  = m0 + l15;

    v8f acc[4] = {};

    for (int kk = 0; kk < 20; ++kk) {
        const int k0 = kk * 32;
        // A fragment (16-bit A 16x32 layout): lane<16 -> K[0:8)+K[16:24),
        // lane>=16 -> K[8:16)+K[24:32); two 16B runs per lane.
        const v4u* ap = (const v4u*)(w4b + (size_t)arow * C5 + k0 + (lhalf ? 8 : 0));
        const v4u a_lo = ap[0];
        const v4u a_hi = ap[2];                 // +16 bf16 elements = +32 bytes
        v8u au;
#pragma unroll
        for (int q = 0; q < 4; ++q) { au[q] = a_lo[q]; au[q + 4] = a_hi[q]; }
        const v16bf afrag = __builtin_bit_cast(v16bf, au);

        const int kb = k0 + (lhalf ? 8 : 0);    // B halves: lanes<16 K[0:8)+K[16:24), >=16 K[8:16)+K[24:32)
#pragma unroll
        for (int n = 0; n < 4; ++n) {
            const int col = 16 * n + l15;
            const v4u* bp = (const v4u*)(&yt[col * YSTR + kb]);
            const v4u b_lo = bp[0];
            const v4u b_hi = bp[2];
            v8u bu;
#pragma unroll
            for (int q = 0; q < 4; ++q) { bu[q] = b_lo[q]; bu[q + 4] = b_hi[q]; }
            const v16bf bfrag = __builtin_bit_cast(v16bf, bu);
            acc[n] = __builtin_amdgcn_wmma_f32_16x16x32_bf16(
                false, afrag, false, bfrag, (short)0, acc[n], false, false);
        }
    }

    // ---- epilogue: + b4 + residual x, store f32 ----
#pragma unroll
    for (int n = 0; n < 4; ++n) {
        const int go_i = i0 + n;                // TW==16 -> tile row index == n
        const int go_j = j0 + l15;
#pragma unroll
        for (int r = 0; r < 8; ++r) {
            const int row = m0 + r + (lhalf ? 8 : 0);
            const size_t oidx = ((size_t)(bb * CC + row) * HH + go_i) * WW + go_j;
            out[oidx] = acc[n][r] + b4[row] + x[oidx];
        }
    }
}

extern "C" void kernel_launch(void* const* d_in, const int* in_sizes, int n_in,
                              void* d_out, int out_size, void* d_ws, size_t ws_size,
                              hipStream_t stream) {
    const float* x  = (const float*)d_in[0];
    const float* w1 = (const float*)d_in[1];
    const float* b1 = (const float*)d_in[2];
    const float* w2 = (const float*)d_in[3];
    const float* b2 = (const float*)d_in[4];
    const float* w3 = (const float*)d_in[5];
    const float* b3 = (const float*)d_in[6];
    const float* w4 = (const float*)d_in[7];
    const float* b4 = (const float*)d_in[8];
    float* out = (float*)d_out;

    __bf16* w4b = (__bf16*)d_ws;                       // 128*640*2 = 160 KB scratch
    const int nW4 = CC * C5;
    lrdw_cvt_w4_bf16<<<(nW4 + 255) / 256, 256, 0, stream>>>(w4, w4b, nW4);

    dim3 grid(WW / TW, HH / TH, BB);                   // 8 x 32 x 8
    lrdw_fused_kernel<<<grid, 256, 0, stream>>>(x, w1, b1, w2, b2, w3, b3, b4, w4b, out);
}
